// StructureEncoder_76699525972514
// MI455X (gfx1250) — matrix operand order repaired and checked by reading, not compiled, not run on hardware
//
#include <hip/hip_runtime.h>
#include <hip/hip_bf16.h>

#define N_NODES 10000
#define E_EDGES 160000
#define NREL 4
#define NHEAD 3
#define NGRAPH 100
#define EDIM 8
#define NEG_SLOPE 0.2f

typedef __attribute__((ext_vector_type(16))) __bf16 v16bf;
typedef __attribute__((ext_vector_type(8)))  __bf16 v8bf;
typedef __attribute__((ext_vector_type(4)))  __bf16 v4bf;
typedef __attribute__((ext_vector_type(8)))  float  v8f;

// ---------------- elementwise helpers ----------------

__global__ void k_f32_to_bf16(const float* __restrict__ src, __bf16* __restrict__ dst, int n) {
  int i = blockIdx.x * blockDim.x + threadIdx.x;
  if (i < n) dst[i] = (__bf16)src[i];
}

__global__ void k_fill_f32(float* __restrict__ p, float v, int n) {
  int i = blockIdx.x * blockDim.x + threadIdx.x;
  if (i < n) p[i] = v;
}

__global__ void k_fill_u32(unsigned* __restrict__ p, unsigned v, int n) {
  int i = blockIdx.x * blockDim.x + threadIdx.x;
  if (i < n) p[i] = v;
}

__global__ void k_bias_fill(float* __restrict__ p, const float* __restrict__ b, int ncols, int n) {
  int i = blockIdx.x * blockDim.x + threadIdx.x;
  if (i < n) p[i] = b[i % ncols];
}

// ---------------- WMMA bf16 GEMM ----------------
// C[m, coff + n] = sum_k A[m,k] * B[z][k,n] (+ bias[n])
// Block: 128 threads = 4 waves. Tile: 16 M x 128 N, K-step 32.
// Each wave computes two 16x16 tiles (n and n+64) -> 2 v_wmma per K-step.
// B tile is staged TRANSPOSED in LDS so each lane's fragment is two aligned
// 16-byte ds_load_b128 reads; global staging uses 8-byte (4 x bf16) loads.
// Requires: M % 16 == 0, Ncols % 128 == 0, K % 32 == 0 (true for all shapes here).
#define BS_STRIDE 40   // bf16 elems per BsT row: 80 B, multiple of 16 B, spreads banks

template <bool HASBIAS>
__global__ __launch_bounds__(128)
void k_gemm_bf16(const __bf16* __restrict__ A, const __bf16* __restrict__ Bm,
                 float* __restrict__ C, const float* __restrict__ bias,
                 int Ncols, int K, int ldc)
{
  const int z = blockIdx.z;
  const __bf16* Bz = Bm + (size_t)z * K * Ncols;
  const int coff = z * Ncols;

  __shared__ __align__(16) __bf16 As[16 * 32];           // 1 KB, [m][k]
  __shared__ __align__(16) __bf16 BsT[128 * BS_STRIDE];  // 10 KB, [n][k]

  const int tid  = threadIdx.x;
  const int wave = tid >> 5;
  const int lane = tid & 31;
  const int row  = lane & 15;   // M row (A frag) / N col (B,D frags) within tile
  const int grp  = lane >> 4;   // K-half selector

  const int m0   = blockIdx.y * 16;
  const int nblk = blockIdx.x * 128;
  const int n0   = wave * 16;        // tile-local column of first 16x16 tile
  const int n1   = n0 + 64;          // second tile

  // staging coordinates (hoisted out of the K loop)
  const int ea   = tid * 4;          // A: element base, 4 bf16 per thread
  const int ma   = ea >> 5;          // A row
  const int ka   = ea & 31;          // A col base

  v8f acc0 = {};
  v8f acc1 = {};

  union V16 { v16bf v; v8bf h[2]; };

  for (int k0 = 0; k0 < K; k0 += 32) {
    // stage A tile (16 x 32): one 8-byte load + one 8-byte LDS store per thread
    *(v4bf*)&As[ea] = *(const v4bf*)&A[(size_t)(m0 + ma) * K + k0 + ka];

    // stage B tile (32 k x 128 n) transposed -> BsT[n][k]
    // 8-byte global loads coalesced along n; scalar b16 scatter into BsT
#pragma unroll
    for (int t = 0; t < 8; ++t) {
      int ch = tid + t * 128;          // 1024 chunks of 4 bf16
      int kk = ch >> 5;                // k row 0..31
      int c4 = (ch & 31) * 4;          // n base 0..124
      v4bf vv = *(const v4bf*)&Bz[(size_t)(k0 + kk) * Ncols + nblk + c4];
      BsT[(c4 + 0) * BS_STRIDE + kk] = vv[0];
      BsT[(c4 + 1) * BS_STRIDE + kk] = vv[1];
      BsT[(c4 + 2) * BS_STRIDE + kk] = vv[2];
      BsT[(c4 + 3) * BS_STRIDE + kk] = vv[3];
    }
    // prefetch next K-step tiles into cache (global_prefetch_b8)
    if (k0 + 32 < K) {
      __builtin_prefetch(&A[(size_t)(m0 + ma) * K + k0 + 32 + ka], 0, 1);
      __builtin_prefetch(&Bz[(size_t)(k0 + 32 + (tid >> 2)) * Ncols + nblk + (tid & 3) * 32], 0, 1);
    }
    __syncthreads();

    // A fragment: lane holds M=row; elem j -> K = grp*16 + j
    V16 af, b0, b1;
    af.h[0] = *(const v8bf*)&As[row * 32 + grp * 16];
    af.h[1] = *(const v8bf*)&As[row * 32 + grp * 16 + 8];
    // B fragment: lane holds N=col; elem j -> K = grp*16 + j (contiguous in BsT row)
    b0.h[0] = *(const v8bf*)&BsT[(n0 + row) * BS_STRIDE + grp * 16];
    b0.h[1] = *(const v8bf*)&BsT[(n0 + row) * BS_STRIDE + grp * 16 + 8];
    b1.h[0] = *(const v8bf*)&BsT[(n1 + row) * BS_STRIDE + grp * 16];
    b1.h[1] = *(const v8bf*)&BsT[(n1 + row) * BS_STRIDE + grp * 16 + 8];

    acc0 = __builtin_amdgcn_wmma_f32_16x16x32_bf16(false, af.v, false, b0.v, (short)0, acc0, false, false);
    acc1 = __builtin_amdgcn_wmma_f32_16x16x32_bf16(false, af.v, false, b1.v, (short)0, acc1, false, false);
    __syncthreads();
  }

  float bv0 = 0.f, bv1 = 0.f;
  if (HASBIAS) {
    bv0 = bias[nblk + n0 + row];
    bv1 = bias[nblk + n1 + row];
  }
  // D layout: lane col = row; VGPR j -> M = grp*8 + j
#pragma unroll
  for (int j = 0; j < 8; ++j) {
    int m = m0 + grp * 8 + j;
    C[(size_t)m * ldc + coff + nblk + n0 + row] = acc0[j] + bv0;
    C[(size_t)m * ldc + coff + nblk + n1 + row] = acc1[j] + bv1;
  }
}

// ---------------- attention projections ----------------

// qn/kn[(n*R + r)*H + h] = sum_o xW[n,r,o] * {q,k}[o,h]
__global__ void k_qk(const float* __restrict__ xW, const float* __restrict__ q,
                     const float* __restrict__ kmat, float* __restrict__ qn,
                     float* __restrict__ kn, int HO, int total)
{
  int i = blockIdx.x * blockDim.x + threadIdx.x;
  if (i >= total) return;
  int h = i % NHEAD;
  int nr = i / NHEAD;
  const float* xb = xW + (size_t)nr * HO;
  float dq = 0.f, dk = 0.f;
  for (int o = 0; o < HO; ++o) {
    float xv = xb[o];
    dq += xv * q[o * NHEAD + h];
    dk += xv * kmat[o * NHEAD + h];
  }
  qn[i] = dq;
  kn[i] = dk;
}

// g[d,h] = sum_o le[d,o] * e[o,h]   (tiny: 24 dots)
__global__ void k_edge_g(const float* __restrict__ le, const float* __restrict__ ev,
                         float* __restrict__ g, int HO)
{
  int t = threadIdx.x;
  if (t >= EDIM * NHEAD) return;
  int d = t / NHEAD, h = t % NHEAD;
  float s = 0.f;
  for (int o = 0; o < HO; ++o) s += le[(size_t)d * HO + o] * ev[o * NHEAD + h];
  g[t] = s;
}

// float atomic max via signed-max / unsigned-min trick (init bits = 0xFF800000 = -inf)
__device__ __forceinline__ void atomicMaxF(float* a, float v) {
  if (v >= 0.f) atomicMax((int*)a, __float_as_int(v));
  else          atomicMin((unsigned int*)a, __float_as_uint(v));
}

__global__ void k_edge_score(const int* __restrict__ src, const int* __restrict__ dst,
                             const int* __restrict__ et, const float* __restrict__ eattr,
                             const float* __restrict__ qn, const float* __restrict__ kn,
                             const float* __restrict__ g, float* __restrict__ alpha,
                             float* __restrict__ amax)
{
  int i = blockIdx.x * blockDim.x + threadIdx.x;
  if (i >= E_EDGES * NHEAD) return;
  int h = i % NHEAD, e = i / NHEAD;
  int r = et[e], s = src[e], d = dst[e];
  float ae = 0.f;
#pragma unroll
  for (int dd = 0; dd < EDIM; ++dd) ae += eattr[e * EDIM + dd] * g[dd * NHEAD + h];
  float a = qn[(d * NREL + r) * NHEAD + h] + kn[(s * NREL + r) * NHEAD + h] + ae;
  a = (a > 0.f) ? a : NEG_SLOPE * a;        // leaky_relu
  alpha[i] = a;
  atomicMaxF(&amax[d * NHEAD + h], a);
}

__global__ void k_edge_exp(const int* __restrict__ dst, float* __restrict__ alpha,
                           const float* __restrict__ amax, float* __restrict__ ssum)
{
  int i = blockIdx.x * blockDim.x + threadIdx.x;
  if (i >= E_EDGES * NHEAD) return;
  int h = i % NHEAD, e = i / NHEAD;
  int d = dst[e];
  float ex = __expf(alpha[i] - amax[d * NHEAD + h]);
  alpha[i] = ex;
  atomicAdd(&ssum[d * NHEAD + h], ex);
}

// aggr[d, o] += (1/H) * sum_h (alpha/sum) * xW[src, et, h*cout + o]
__global__ void k_edge_aggr(const int* __restrict__ src, const int* __restrict__ dst,
                            const int* __restrict__ et, const float* __restrict__ alpha,
                            const float* __restrict__ ssum, const float* __restrict__ xW,
                            float* __restrict__ aggr, int cout, int HO)
{
  int i = blockIdx.x * blockDim.x + threadIdx.x;
  if (i >= E_EDGES * cout) return;
  int o = i % cout, e = i / cout;
  int r = et[e], s = src[e], d = dst[e];
  const float* xb = xW + ((size_t)s * NREL + r) * HO;
  float acc = 0.f;
#pragma unroll
  for (int h = 0; h < NHEAD; ++h) {
    float w = alpha[e * NHEAD + h] / fmaxf(ssum[d * NHEAD + h], 1e-16f);
    acc += w * xb[h * cout + o];
  }
  atomicAdd(&aggr[(size_t)d * cout + o], acc * (1.f / NHEAD));
}

// ---------------- readout ----------------

__global__ void k_hole(const int* __restrict__ batch, int* __restrict__ hole, int n) {
  int i = blockIdx.x * blockDim.x + threadIdx.x;
  if (i < n) atomicMin(&hole[batch[i]], i);
}

__global__ void k_softmax(const float* __restrict__ h, const int* __restrict__ hole,
                          float* __restrict__ out)
{
  __shared__ float red[256];
  int b = blockIdx.x, t = threadIdx.x;
  int n = hole[b];
  float v = h[(size_t)n * 256 + t];
  red[t] = v;
  __syncthreads();
  for (int s = 128; s > 0; s >>= 1) { if (t < s) red[t] = fmaxf(red[t], red[t + s]); __syncthreads(); }
  float mx = red[0];
  __syncthreads();
  float ex = __expf(v - mx);
  red[t] = ex;
  __syncthreads();
  for (int s = 128; s > 0; s >>= 1) { if (t < s) red[t] += red[t + s]; __syncthreads(); }
  out[(size_t)b * 256 + t] = ex / red[0];
}

// ---------------- orchestration ----------------

extern "C" void kernel_launch(void* const* d_in, const int* in_sizes, int n_in,
                              void* d_out, int out_size, void* d_ws, size_t ws_size,
                              hipStream_t stream)
{
  (void)in_sizes; (void)n_in; (void)out_size; (void)ws_size;

  const float* x     = (const float*)d_in[0];
  const int*   eidx  = (const int*)d_in[1];
  const int*   etype = (const int*)d_in[2];
  const float* eattr = (const float*)d_in[3];
  const int*   batch = (const int*)d_in[4];
  const int* src = eidx;             // edge_index[0]
  const int* dst = eidx + E_EDGES;   // edge_index[1]

  const float *W[4], *q[4], *kmat[4], *le[4], *ev[4], *bb[4], *lw[4], *lb[4];
  for (int i = 0; i < 4; ++i) {
    int base = 5 + i * 6;
    W[i]    = (const float*)d_in[base + 0];
    q[i]    = (const float*)d_in[base + 1];
    kmat[i] = (const float*)d_in[base + 2];
    le[i]   = (const float*)d_in[base + 3];
    ev[i]   = (const float*)d_in[base + 4];
    bb[i]   = (const float*)d_in[base + 5];
  }
  for (int i = 0; i < 4; ++i) {
    lw[i] = (const float*)d_in[29 + i * 2];
    lb[i] = (const float*)d_in[30 + i * 2];
  }

  static const int CIN[4]  = {768, 256, 256, 256};
  static const int COUT[4] = {128, 384, 128, 128};

  // workspace bump allocator (256B aligned)
  char* p = (char*)d_ws;
  auto alloc = [&](size_t bytes) -> void* {
    void* r = (void*)p;
    p += (bytes + 255) & ~(size_t)255;
    return r;
  };
  float*  xW    = (float*) alloc((size_t)N_NODES * NREL * 1152 * sizeof(float)); // 184 MB max
  __bf16* abf   = (__bf16*)alloc((size_t)N_NODES * 768 * sizeof(__bf16));
  __bf16* wbf   = (__bf16*)alloc((size_t)NREL * 768 * 384 * sizeof(__bf16));     // covers all W / lw
  float*  qn    = (float*) alloc((size_t)N_NODES * NREL * NHEAD * sizeof(float));
  float*  kn    = (float*) alloc((size_t)N_NODES * NREL * NHEAD * sizeof(float));
  float*  gsc   = (float*) alloc(EDIM * NHEAD * sizeof(float));
  float*  alpha = (float*) alloc((size_t)E_EDGES * NHEAD * sizeof(float));
  float*  amax  = (float*) alloc((size_t)N_NODES * NHEAD * sizeof(float));
  float*  ssum  = (float*) alloc((size_t)N_NODES * NHEAD * sizeof(float));
  float*  aggr  = (float*) alloc((size_t)N_NODES * 384 * sizeof(float));
  float*  hbuf  = (float*) alloc((size_t)N_NODES * 256 * sizeof(float));
  int*    hole  = (int*)   alloc(NGRAPH * sizeof(int));

  const int TB = 256;
  auto nb = [](long long n, int tb) { return (unsigned)((n + tb - 1) / tb); };

  for (int l = 0; l < 4; ++l) {
    const int cin = CIN[l], cout = COUT[l], HO = NHEAD * cout;
    const float* hin = (l == 0) ? x : hbuf;

    // bf16 conversions for WMMA
    long long nA = (long long)N_NODES * cin;
    k_f32_to_bf16<<<nb(nA, TB), TB, 0, stream>>>(hin, abf, (int)nA);
    long long nW = (long long)NREL * cin * HO;
    k_f32_to_bf16<<<nb(nW, TB), TB, 0, stream>>>(W[l], wbf, (int)nW);

    // xW[n, r, :] = h @ W[r]   (batched over relations via blockIdx.z)
    dim3 gg(HO / 128, N_NODES / 16, NREL);
    k_gemm_bf16<false><<<gg, 128, 0, stream>>>(abf, wbf, xW, nullptr, HO, cin, NREL * HO);

    // attention projections
    long long nqk = (long long)N_NODES * NREL * NHEAD;
    k_qk<<<nb(nqk, TB), TB, 0, stream>>>(xW, q[l], kmat[l], qn, kn, HO, (int)nqk);
    k_edge_g<<<1, 32, 0, stream>>>(le[l], ev[l], gsc, HO);

    // init segment buffers: amax=-inf, ssum=0, aggr=bias (head-mean bias folded in)
    long long nnh = (long long)N_NODES * NHEAD;
    k_fill_u32<<<nb(nnh, TB), TB, 0, stream>>>((unsigned*)amax, 0xFF800000u, (int)nnh);
    k_fill_f32<<<nb(nnh, TB), TB, 0, stream>>>(ssum, 0.f, (int)nnh);
    long long nag = (long long)N_NODES * cout;
    k_bias_fill<<<nb(nag, TB), TB, 0, stream>>>(aggr, bb[l], cout, (int)nag);

    // edge phase: score + segmax, exp + segsum, weighted scatter
    long long neh = (long long)E_EDGES * NHEAD;
    k_edge_score<<<nb(neh, TB), TB, 0, stream>>>(src, dst, etype, eattr, qn, kn, gsc, alpha, amax);
    k_edge_exp<<<nb(neh, TB), TB, 0, stream>>>(dst, alpha, amax, ssum);
    long long neo = (long long)E_EDGES * cout;
    k_edge_aggr<<<nb(neo, TB), TB, 0, stream>>>(src, dst, etype, alpha, ssum, xW, aggr, cout, HO);

    // linear: hbuf = aggr @ lw + lb  (WMMA, bias fused in epilogue)
    k_f32_to_bf16<<<nb(nag, TB), TB, 0, stream>>>(aggr, abf, (int)nag);
    long long nlw = (long long)cout * 256;
    k_f32_to_bf16<<<nb(nlw, TB), TB, 0, stream>>>(lw[l], wbf, (int)nlw);
    dim3 gl(256 / 128, N_NODES / 16, 1);
    k_gemm_bf16<true><<<gl, 128, 0, stream>>>(abf, wbf, hbuf, lb[l], 256, cout, 256);
  }

  // hole node per graph (= segment_min of node index) + row softmax
  k_fill_u32<<<nb(NGRAPH, TB), TB, 0, stream>>>((unsigned*)hole, 0x7FFFFFFFu, NGRAPH);
  k_hole<<<nb(N_NODES, TB), TB, 0, stream>>>(batch, hole, N_NODES);
  k_softmax<<<NGRAPH, 256, 0, stream>>>(hbuf, hole, (float*)d_out);
}
